// NeRFRenderer_40922448396820
// MI455X (gfx1250) — matrix operand order repaired and checked by reading, not compiled, not run on hardware
//
#include <hip/hip_runtime.h>

#define NRAYS 8192
#define TS    256
#define WAVES 4
#define BLOCK (WAVES * 32)

typedef __attribute__((ext_vector_type(16))) _Float16 v16h;
typedef __attribute__((ext_vector_type(8)))  _Float16 v8h;
typedef __attribute__((ext_vector_type(8)))  float    v8f;
typedef __attribute__((ext_vector_type(4)))  unsigned int u32x4;
typedef __attribute__((ext_vector_type(8)))  int i32x8;
typedef __attribute__((ext_vector_type(4)))  int i32x4;

#if defined(__HIP_DEVICE_COMPILE__) && __has_builtin(__builtin_amdgcn_tensor_load_to_lds)
#define USE_TDM 1
#else
#define USE_TDM 0
#endif

__device__ __forceinline__ v16h cat16(v8h lo, v8h hi) {
  return __builtin_shufflevector(lo, hi, 0,1,2,3,4,5,6,7,8,9,10,11,12,13,14,15);
}

// A fragment (16x32 f16, row-major staging in LDS, stride in halves).
// Lane L: row m=L&15; lanes 0-15 hold K in [kh,kh+8) U [kh+16,kh+24) with kh=0,
// lanes 16-31 with kh=8 (CDNA5 ISA 7.12.2 16-bit A layout).
__device__ __forceinline__ v16h ldA(const _Float16* base, int strideH, int lane, int koff) {
  int m  = lane & 15;
  int kh = (lane >> 4) << 3;
  const _Float16* p = base + m * strideH + koff + kh;
  return cat16(*(const v8h*)p, *(const v8h*)(p + 16));
}

// B fragment (32x16 f16). Weights stored transposed (N-major, K-contiguous) in LDS.
// Lane L: col n=L&15; K in [kb,kb+8) U [kb+16,kb+24) with kb=(L/16)*8 (+ k-step offset).
__device__ __forceinline__ v16h ldB(const _Float16* baseT, int strideH, int lane, int koff) {
  int n  = lane & 15;
  int kb = ((lane >> 4) << 3) + koff;
  const _Float16* p = baseT + n * strideH + kb;
  return cat16(*(const v8h*)p, *(const v8h*)(p + 16));
}

__device__ __forceinline__ v8f wmma16(v16h a, v16h b, v8f c) {
  return __builtin_amdgcn_wmma_f32_16x16x32_f16(false, a, false, b, (short)0, c, false, false);
}

#if USE_TDM
// 1-D TDM copy: nelems f32 from global -> LDS (CDNA5 ISA ch.8 D# layout).
__device__ __forceinline__ void tdm_load_1d(unsigned lds_addr, const void* gsrc, unsigned nelems) {
  unsigned long long ga = (unsigned long long)(uintptr_t)gsrc;
  u32x4 g0;
  g0.x = 1u;                                                   // count=1, user mode
  g0.y = lds_addr;                                             // LDS byte address
  g0.z = (unsigned)ga;                                         // global_addr[31:0]
  g0.w = (unsigned)((ga >> 32) & 0x01ffffffull) | (2u << 30);  // global_addr[56:32] | type=2
  i32x8 g1;
  g1[0] = (int)(2u << 16);                                     // data_size = 4B
  g1[1] = (int)((nelems & 0xffffu) << 16);                     // tensor_dim0[15:0]
  g1[2] = (int)(((nelems >> 16) & 0xffffu) | (1u << 16));      // tensor_dim0[31:16] | tensor_dim1=1
  g1[3] = (int)((nelems & 0xffffu) << 16);                     // tile_dim0 = nelems
  g1[4] = 1;                                                   // tile_dim1 = 1
  g1[5] = (int)nelems;                                         // tensor_dim0_stride
  g1[6] = 0;
  g1[7] = 0;
  i32x4 z4 = {0, 0, 0, 0};
#if __clang_major__ >= 23
  i32x8 z8 = {0, 0, 0, 0, 0, 0, 0, 0};
  __builtin_amdgcn_tensor_load_to_lds(g0, g1, z4, z4, z8, 0);
#else
  __builtin_amdgcn_tensor_load_to_lds(g0, g1, z4, z4, 0);
#endif
}
#endif

__global__ __launch_bounds__(BLOCK) void nerf_render_kernel(
    const float* __restrict__ rays_o, const float* __restrict__ rays_d,
    const float* __restrict__ dnorms,
    const float* __restrict__ Wd1, const float* __restrict__ bd1,
    const float* __restrict__ Wd2, const float* __restrict__ bd2,
    const float* __restrict__ Wc1, const float* __restrict__ bc1,
    const float* __restrict__ Wc2, const float* __restrict__ bc2,
    const float* __restrict__ Wk1, const float* __restrict__ bk1,
    const float* __restrict__ Wk2, const float* __restrict__ bk2,
    float* __restrict__ out)
{
  // ---- weights (shared by all waves), f16 transposed, K-contiguous ----
  __shared__ _Float16 sWd1T[64 * 32];   // [n][k], k<3 valid
  __shared__ _Float16 sWd2T[16 * 64];   // [n][k]
  __shared__ _Float16 sWc1T[64 * 32];   // [n][k], k<18 valid
  __shared__ _Float16 sWc2T[16 * 64];   // [n][k], n<3 valid
  __shared__ _Float16 sWk1T[64 * 32];   // [n][k], k<16 valid
  __shared__ _Float16 sWk2T[32 * 64];   // [n][k]
  __shared__ float sbd1[64], sbd2[16], sbc1[64], sbc2[16], sbk1[64], sbk2[32];
  // ---- per-wave staging (private slices, same-wave LDS ordering only) ----
  __shared__ _Float16 sXin[WAVES * 32 * 32];  // xyz, K pad 32
  __shared__ _Float16 sCin[WAVES * 32 * 32];  // [dir(3), geo(15)], K pad 32
  __shared__ _Float16 sKin[WAVES * 32 * 32];  // [geo(15), sigma], K pad 32
  __shared__ _Float16 sHH [WAVES * 32 * 64];  // hidden (64), reused by all 3 MLPs
  __shared__ float sSig[WAVES * 32];
  __shared__ float sWl [WAVES * 32];
  __shared__ float sRgb[WAVES * 32 * 4];

  const int tid = threadIdx.x;

  // -------- async-tensor (TDM) preload of raw f32 weights into LDS --------
#if USE_TDM
  __shared__ float stagW[192 + 1024 + 1152 + 192 + 1024 + 2048];  // 5632 f32
  if (tid < 32) {  // one wave issues; TDM ignores EXEC, tracked by TENSORcnt
    tdm_load_1d((unsigned)(uintptr_t)&stagW[0],    Wd1, 192);
    tdm_load_1d((unsigned)(uintptr_t)&stagW[192],  Wd2, 1024);
    tdm_load_1d((unsigned)(uintptr_t)&stagW[1216], Wc1, 1152);
    tdm_load_1d((unsigned)(uintptr_t)&stagW[2368], Wc2, 192);
    tdm_load_1d((unsigned)(uintptr_t)&stagW[2560], Wk1, 1024);
    tdm_load_1d((unsigned)(uintptr_t)&stagW[3584], Wk2, 2048);
    __builtin_amdgcn_s_wait_tensorcnt((short)0);
  }
  __syncthreads();
  const float* pWd1 = &stagW[0];
  const float* pWd2 = &stagW[192];
  const float* pWc1 = &stagW[1216];
  const float* pWc2 = &stagW[2368];
  const float* pWk1 = &stagW[2560];
  const float* pWk2 = &stagW[3584];
#else
  const float* pWd1 = Wd1; const float* pWd2 = Wd2; const float* pWc1 = Wc1;
  const float* pWc2 = Wc2; const float* pWk1 = Wk1; const float* pWk2 = Wk2;
#endif

  // -------- block init: transpose + f16-convert weights, zero staging pads --------
  for (int i = tid; i < 64 * 32; i += BLOCK) {
    int n = i >> 5, k = i & 31;
    sWd1T[i] = (k < 3)  ? (_Float16)pWd1[k * 64 + n] : (_Float16)0.f;
    sWc1T[i] = (k < 18) ? (_Float16)pWc1[k * 64 + n] : (_Float16)0.f;
    sWk1T[i] = (k < 16) ? (_Float16)pWk1[k * 64 + n] : (_Float16)0.f;
  }
  for (int i = tid; i < 16 * 64; i += BLOCK) {
    int n = i >> 6, k = i & 63;
    sWd2T[i] = (_Float16)pWd2[k * 16 + n];
    sWc2T[i] = (n < 3) ? (_Float16)pWc2[k * 3 + n] : (_Float16)0.f;
  }
  for (int i = tid; i < 32 * 64; i += BLOCK) {
    int n = i >> 6, k = i & 63;
    sWk2T[i] = (_Float16)pWk2[k * 32 + n];
  }
  for (int i = tid; i < 64; i += BLOCK) { sbd1[i] = bd1[i]; sbc1[i] = bc1[i]; sbk1[i] = bk1[i]; }
  for (int i = tid; i < 32; i += BLOCK) { sbk2[i] = bk2[i]; }
  for (int i = tid; i < 16; i += BLOCK) { sbd2[i] = bd2[i]; sbc2[i] = (i < 3) ? bc2[i] : 0.f; }
  for (int i = tid; i < WAVES * 32 * 32; i += BLOCK) {
    sXin[i] = (_Float16)0.f; sCin[i] = (_Float16)0.f; sKin[i] = (_Float16)0.f;
  }
  __syncthreads();

  const int wave = tid >> 5, lane = tid & 31;
  _Float16* xin  = &sXin[wave * 1024];
  _Float16* cin  = &sCin[wave * 1024];
  _Float16* kin  = &sKin[wave * 1024];
  _Float16* hbuf = &sHH [wave * 2048];
  float*    sig  = &sSig[wave * 32];
  float*    wls  = &sWl [wave * 32];
  float*    rgbt = &sRgb[wave * 128];

  float* out_img   = out;
  float* out_depth = out + 3 * NRAYS;
  float* out_var   = out + 4 * NRAYS;
  float* out_coord = out + 5 * NRAYS;
  float* out_rgbs  = out + 8 * NRAYS;
  float* out_clip  = out + 8 * NRAYS + 3 * NRAYS * TS;

  const int n     = lane & 15;         // WMMA C-tile column for this lane
  const int rbase = (lane >> 4) << 3;  // C-tile row base (0 or 8)

  for (int ray = blockIdx.x * WAVES + wave; ray < NRAYS; ray += (int)gridDim.x * WAVES) {
    __builtin_prefetch(rays_o + (ray + (int)gridDim.x * WAVES) * 3, 0, 0);
    const float ox = rays_o[ray * 3 + 0], oy = rays_o[ray * 3 + 1], oz = rays_o[ray * 3 + 2];
    const float dx = rays_d[ray * 3 + 0], dy = rays_d[ray * 3 + 1], dz = rays_d[ray * 3 + 2];
    const float dn = dnorms[ray];
    const float ivx = 1.f / dx, ivy = 1.f / dy, ivz = 1.f / dz;
    const float t1x = (-1.f - ox) * ivx, t2x = (1.f - ox) * ivx;
    const float t1y = (-1.f - oy) * ivy, t2y = (1.f - oy) * ivy;
    const float t1z = (-1.f - oz) * ivz, t2z = (1.f - oz) * ivz;
    float nearv = fmaxf(fmaxf(fminf(t1x, t2x), fminf(t1y, t2y)), fminf(t1z, t2z));
    nearv = fmaxf(nearv, 0.2f);
    float farv = fminf(fminf(fmaxf(t1x, t2x), fmaxf(t1y, t2y)), fmaxf(t1z, t2z));
    farv = fmaxf(farv, nearv + 1e-4f);
    const float dstep = (farv - nearv) / (float)(TS - 1);   // uniform linspace spacing
    const float sdist = (farv - nearv) / (float)TS;

    // dir columns of color-MLP input (constant over the ray), row = lane
    cin[lane * 32 + 0] = (_Float16)dx;
    cin[lane * 32 + 1] = (_Float16)dy;
    cin[lane * 32 + 2] = (_Float16)dz;

    float carry = 1.f;
    float wsum = 0.f, dsum = 0.f, d2s = 0.f;
    float cxa = 0.f, cya = 0.f, cza = 0.f;
    float ir = 0.f, ig = 0.f, ib = 0.f;
    float acl0 = 0.f, acl1 = 0.f;

    for (int c = 0; c < TS / 32; ++c) {
      const int   s  = c * 32 + lane;
      const float zf = nearv + dstep * (float)s;
      const float px = fminf(fmaxf(ox + dx * zf, -1.f), 1.f);
      const float py = fminf(fmaxf(oy + dy * zf, -1.f), 1.f);
      const float pz = fminf(fmaxf(oz + dz * zf, -1.f), 1.f);
      xin[lane * 32 + 0] = (_Float16)px;
      xin[lane * 32 + 1] = (_Float16)py;
      xin[lane * 32 + 2] = (_Float16)pz;

      // ---------------- density MLP: 3 -> 64 (relu) -> 16 ----------------
      {
        v16h bf[4];
        #pragma unroll
        for (int nt = 0; nt < 4; ++nt) bf[nt] = ldB(sWd1T + nt * 16 * 32, 32, lane, 0);
        #pragma unroll
        for (int mt = 0; mt < 2; ++mt) {
          v16h a = ldA(xin + mt * 16 * 32, 32, lane, 0);
          int  mb = mt * 16 + rbase;
          #pragma unroll
          for (int nt = 0; nt < 4; ++nt) {
            v8f cc = {0.f, 0.f, 0.f, 0.f, 0.f, 0.f, 0.f, 0.f};
            cc = wmma16(a, bf[nt], cc);
            float bias = sbd1[nt * 16 + n];
            #pragma unroll
            for (int r = 0; r < 8; ++r) {
              float v = fmaxf(cc[r] + bias, 0.f);
              hbuf[(mb + r) * 64 + nt * 16 + n] = (_Float16)v;
            }
          }
        }
      }
      {
        v16h b0 = ldB(sWd2T, 64, lane, 0);
        v16h b1 = ldB(sWd2T, 64, lane, 32);
        #pragma unroll
        for (int mt = 0; mt < 2; ++mt) {
          int  mb = mt * 16 + rbase;
          v16h a0 = ldA(hbuf + mt * 16 * 64, 64, lane, 0);
          v16h a1 = ldA(hbuf + mt * 16 * 64, 64, lane, 32);
          v8f  cc = {0.f, 0.f, 0.f, 0.f, 0.f, 0.f, 0.f, 0.f};
          cc = wmma16(a0, b0, cc);
          cc = wmma16(a1, b1, cc);
          float bias = sbd2[n];
          if (n == 0) {  // sigma column
            #pragma unroll
            for (int r = 0; r < 8; ++r) {
              float sgm = __expf(cc[r] + bias);
              sig[mb + r] = sgm;
              kin[(mb + r) * 32 + 15] = (_Float16)sgm;
            }
          } else {       // geo_feat columns 1..15
            #pragma unroll
            for (int r = 0; r < 8; ++r) {
              float g = cc[r] + bias;
              cin[(mb + r) * 32 + 2 + n] = (_Float16)g;
              kin[(mb + r) * 32 + n - 1] = (_Float16)g;
            }
          }
        }
      }

      // ---------------- compositing (wave32 prefix-product scan) ----------------
      float sgm   = sig[lane];
      float delta = (s == TS - 1) ? sdist : dstep;
      float alpha = 1.f - __expf(-delta * sgm);
      float incl  = 1.f - alpha + 1e-15f;
      #pragma unroll
      for (int off = 1; off < 32; off <<= 1) {
        float t = __shfl_up(incl, off);
        if (lane >= off) incl *= t;
      }
      float excl = __shfl_up(incl, 1);
      if (lane == 0) excl = 1.f;
      float trans = carry * excl;
      float w = alpha * trans;
      carry *= __shfl(incl, 31);
      bool  msk = (w > 1e-4f);
      float wm  = msk ? w : 0.f;
      wls[lane] = wm;
      wsum += wm; dsum += wm * zf; d2s += wm * zf * zf;
      cxa += wm * px; cya += wm * py; cza += wm * pz;

      // ---------------- color MLP: 18 -> 64 (relu) -> 3 (sigmoid) ----------------
      {
        v16h bf[4];
        #pragma unroll
        for (int nt = 0; nt < 4; ++nt) bf[nt] = ldB(sWc1T + nt * 16 * 32, 32, lane, 0);
        #pragma unroll
        for (int mt = 0; mt < 2; ++mt) {
          v16h a = ldA(cin + mt * 16 * 32, 32, lane, 0);
          int  mb = mt * 16 + rbase;
          #pragma unroll
          for (int nt = 0; nt < 4; ++nt) {
            v8f cc = {0.f, 0.f, 0.f, 0.f, 0.f, 0.f, 0.f, 0.f};
            cc = wmma16(a, bf[nt], cc);
            float bias = sbc1[nt * 16 + n];
            #pragma unroll
            for (int r = 0; r < 8; ++r) {
              float v = fmaxf(cc[r] + bias, 0.f);
              hbuf[(mb + r) * 64 + nt * 16 + n] = (_Float16)v;
            }
          }
        }
      }
      {
        v16h b0 = ldB(sWc2T, 64, lane, 0);
        v16h b1 = ldB(sWc2T, 64, lane, 32);
        #pragma unroll
        for (int mt = 0; mt < 2; ++mt) {
          int  mb = mt * 16 + rbase;
          v16h a0 = ldA(hbuf + mt * 16 * 64, 64, lane, 0);
          v16h a1 = ldA(hbuf + mt * 16 * 64, 64, lane, 32);
          v8f  cc = {0.f, 0.f, 0.f, 0.f, 0.f, 0.f, 0.f, 0.f};
          cc = wmma16(a0, b0, cc);
          cc = wmma16(a1, b1, cc);
          if (n < 3) {
            float bias = sbc2[n];
            #pragma unroll
            for (int r = 0; r < 8; ++r) rgbt[(mb + r) * 4 + n] = cc[r] + bias;
          }
        }
      }
      float pr = rgbt[lane * 4 + 0];
      float pg = rgbt[lane * 4 + 1];
      float pb = rgbt[lane * 4 + 2];
      float rr = msk ? 1.f / (1.f + __expf(-pr)) : 0.f;
      float rg = msk ? 1.f / (1.f + __expf(-pg)) : 0.f;
      float rb = msk ? 1.f / (1.f + __expf(-pb)) : 0.f;
      int ridx = (ray * TS + s) * 3;
      out_rgbs[ridx + 0] = rr;
      out_rgbs[ridx + 1] = rg;
      out_rgbs[ridx + 2] = rb;
      ir += wm * rr; ig += wm * rg; ib += wm * rb;

      // ---------------- clip MLP: 16 -> 64 (relu) -> 32 ----------------
      {
        v16h bf[4];
        #pragma unroll
        for (int nt = 0; nt < 4; ++nt) bf[nt] = ldB(sWk1T + nt * 16 * 32, 32, lane, 0);
        #pragma unroll
        for (int mt = 0; mt < 2; ++mt) {
          v16h a = ldA(kin + mt * 16 * 32, 32, lane, 0);
          int  mb = mt * 16 + rbase;
          #pragma unroll
          for (int nt = 0; nt < 4; ++nt) {
            v8f cc = {0.f, 0.f, 0.f, 0.f, 0.f, 0.f, 0.f, 0.f};
            cc = wmma16(a, bf[nt], cc);
            float bias = sbk1[nt * 16 + n];
            #pragma unroll
            for (int r = 0; r < 8; ++r) {
              float v = fmaxf(cc[r] + bias, 0.f);
              hbuf[(mb + r) * 64 + nt * 16 + n] = (_Float16)v;
            }
          }
        }
      }
      {
        v16h bk[2][2];
        #pragma unroll
        for (int nt = 0; nt < 2; ++nt) {
          bk[nt][0] = ldB(sWk2T + nt * 16 * 64, 64, lane, 0);
          bk[nt][1] = ldB(sWk2T + nt * 16 * 64, 64, lane, 32);
        }
        #pragma unroll
        for (int mt = 0; mt < 2; ++mt) {
          int  mb = mt * 16 + rbase;
          v16h a0 = ldA(hbuf + mt * 16 * 64, 64, lane, 0);
          v16h a1 = ldA(hbuf + mt * 16 * 64, 64, lane, 32);
          float wr[8];
          #pragma unroll
          for (int r = 0; r < 8; ++r) wr[r] = wls[mb + r];
          #pragma unroll
          for (int nt = 0; nt < 2; ++nt) {
            v8f cc = {0.f, 0.f, 0.f, 0.f, 0.f, 0.f, 0.f, 0.f};
            cc = wmma16(a0, bk[nt][0], cc);
            cc = wmma16(a1, bk[nt][1], cc);
            float bias = sbk2[nt * 16 + n];
            float acc = 0.f;
            #pragma unroll
            for (int r = 0; r < 8; ++r) acc += wr[r] * (cc[r] + bias);
            if (nt == 0) acl0 += acc; else acl1 += acc;
          }
        }
      }
    }  // chunk loop

    // ---------------- per-ray reductions & stores ----------------
    #pragma unroll
    for (int off = 16; off >= 1; off >>= 1) {
      wsum += __shfl_xor(wsum, off);
      dsum += __shfl_xor(dsum, off);
      d2s  += __shfl_xor(d2s,  off);
      cxa  += __shfl_xor(cxa,  off);
      cya  += __shfl_xor(cya,  off);
      cza  += __shfl_xor(cza,  off);
      ir   += __shfl_xor(ir,   off);
      ig   += __shfl_xor(ig,   off);
      ib   += __shfl_xor(ib,   off);
    }
    acl0 += __shfl_xor(acl0, 16);  // combine row halves of the C tiles
    acl1 += __shfl_xor(acl1, 16);
    if (lane < 16) {
      out_clip[ray * 32 + lane]      = acl0;
      out_clip[ray * 32 + 16 + lane] = acl1;
    }
    if (lane == 0) {
      float depth = dsum / dn;
      out_depth[ray] = depth;
      out_var[ray]   = wsum * depth * depth - 2.f * depth * (dsum / dn) + d2s / (dn * dn);
      float bg = 1.f - wsum;
      out_img[ray * 3 + 0] = ir + bg;
      out_img[ray * 3 + 1] = ig + bg;
      out_img[ray * 3 + 2] = ib + bg;
      out_coord[ray * 3 + 0] = cxa;
      out_coord[ray * 3 + 1] = cya;
      out_coord[ray * 3 + 2] = cza;
    }
  }  // ray loop
}

extern "C" void kernel_launch(void* const* d_in, const int* in_sizes, int n_in,
                              void* d_out, int out_size, void* d_ws, size_t ws_size,
                              hipStream_t stream) {
  (void)in_sizes; (void)n_in; (void)out_size; (void)d_ws; (void)ws_size;
  const float* rays_o = (const float*)d_in[0];
  const float* rays_d = (const float*)d_in[1];
  const float* dnorms = (const float*)d_in[2];
  const float* Wd1 = (const float*)d_in[3];
  const float* bd1 = (const float*)d_in[4];
  const float* Wd2 = (const float*)d_in[5];
  const float* bd2 = (const float*)d_in[6];
  const float* Wc1 = (const float*)d_in[7];
  const float* bc1 = (const float*)d_in[8];
  const float* Wc2 = (const float*)d_in[9];
  const float* bc2 = (const float*)d_in[10];
  const float* Wk1 = (const float*)d_in[11];
  const float* bk1 = (const float*)d_in[12];
  const float* Wk2 = (const float*)d_in[13];
  const float* bk2 = (const float*)d_in[14];
  nerf_render_kernel<<<NRAYS / WAVES, BLOCK, 0, stream>>>(
      rays_o, rays_d, dnorms, Wd1, bd1, Wd2, bd2, Wc1, bc1, Wc2, bc2,
      Wk1, bk1, Wk2, bk2, (float*)d_out);
}
